// LoLGNN_72937134621197
// MI455X (gfx1250) — matrix operand order repaired and checked by reading, not compiled, not run on hardware
//
#include <hip/hip_runtime.h>

typedef __attribute__((ext_vector_type(16))) _Float16 v16h;
typedef __attribute__((ext_vector_type(8)))  float    v8f;

#define NUM_GRAPHS 4096
#define NP 40960
#define HPP 20
#define NH 819200
#define NSEG 8192

__device__ __forceinline__ int clampi(int v, int lo, int hi) {
    return v < lo ? lo : (v > hi ? hi : v);
}

// ---------------------------------------------------------------------------
// 1) Role stats + global win sum (shared-mem partials -> global atomics)
// acc layout: [0..9] cnt, [10..119] sum, [120..229] sumsq, [230] win_sum
// ---------------------------------------------------------------------------
__global__ void k_role_stats(const float* __restrict__ hx, float* __restrict__ acc) {
    __shared__ float s[232];
    int tid = threadIdx.x;
    for (int i = tid; i < 232; i += blockDim.x) s[i] = 0.f;
    __syncthreads();
    int h = blockIdx.x * blockDim.x + tid;
    if (h < NH) {
        const float* r = hx + (size_t)h * 15;
        int role = (int)r[3]; if (role < 0) role = 0;
        atomicAdd(&s[230], fminf(fmaxf(r[1], 0.f), 1.f));
        if (role < 10) {
            atomicAdd(&s[role], 1.f);
            #pragma unroll
            for (int j = 0; j < 11; ++j) {
                float v = r[4 + j];
                atomicAdd(&s[10 + role * 11 + j], v);
                atomicAdd(&s[120 + role * 11 + j], v * v);
            }
        }
    }
    __syncthreads();
    for (int i = tid; i < 231; i += blockDim.x)
        if (s[i] != 0.f) atomicAdd(&acc[i], s[i]);
}

// stats layout: [0..109] mean, [110..219] inv_std, [220] global_wr
__global__ void k_finalize_stats(const float* __restrict__ acc, float* __restrict__ stats) {
    int i = threadIdx.x;
    if (i < 110) {
        int role = i / 11;
        float c = fmaxf(acc[role], 1.f);
        float mean = acc[10 + i] / c;
        float var  = acc[120 + i] / c - mean * mean;
        float sd   = fmaxf(sqrtf(fmaxf(var, 0.f)), 1e-6f);
        stats[i]       = mean;
        stats[110 + i] = 1.f / sd;
    }
    if (i == 110) stats[220] = acc[230] / (float)NH;
}

// ---------------------------------------------------------------------------
// 2) Per-player history aggregation: mean 44-dim feature (f16, padded to 64),
//    games count and smoothed champ winrate.
// ---------------------------------------------------------------------------
__global__ void k_hist_agg(const float* __restrict__ hx, const float* __restrict__ px,
                           const float* __restrict__ E0, const float* __restrict__ E3,
                           const float* __restrict__ stats,
                           _Float16* __restrict__ mfeat, float* __restrict__ games,
                           float* __restrict__ cwr) {
    int p = blockIdx.x * blockDim.x + threadIdx.x;
    if (p >= NP) return;
    float fc[24]; float fe0[16]; float fe3[4];
    #pragma unroll
    for (int j = 0; j < 24; ++j) fc[j] = 0.f;
    #pragma unroll
    for (int j = 0; j < 16; ++j) fe0[j] = 0.f;
    #pragma unroll
    for (int j = 0; j < 4; ++j) fe3[j] = 0.f;
    int pch = (int)px[(size_t)p * 12 + 1]; if (pch < 0) pch = 0;
    float g = 0.f, wsum = 0.f;
    float gwr = stats[220];
    for (int i = 0; i < HPP; ++i) {
        const float* r = hx + (size_t)(p * HPP + i) * 15;
        int champ = (int)r[0]; if (champ < 0) champ = 0;
        int role  = (int)r[3]; if (role < 0) role = 0;
        int rc = role > 9 ? 9 : role;
        fc[0] += r[1];
        fc[1] += r[2];
        #pragma unroll
        for (int j = 0; j < 11; ++j) {
            float v = r[4 + j];
            fc[2 + j]  += v;
            fc[13 + j] += (v - stats[rc * 11 + j]) * stats[110 + rc * 11 + j];
        }
        int ce = champ > 1999 ? 1999 : champ;
        #pragma unroll
        for (int j = 0; j < 16; ++j) fe0[j] += E0[(size_t)ce * 16 + j];
        #pragma unroll
        for (int j = 0; j < 4; ++j) fe3[j] += E3[rc * 4 + j];
        float m = (champ == pch) ? 1.f : 0.f;
        g += m;
        wsum += m * fminf(fmaxf(r[1], 0.f), 1.f);
    }
    _Float16* o = mfeat + (size_t)p * 64;
    const float inv = 1.f / (float)HPP;
    #pragma unroll
    for (int j = 0; j < 24; ++j) o[j] = (_Float16)(fc[j] * inv);
    #pragma unroll
    for (int j = 0; j < 16; ++j) o[24 + j] = (_Float16)(fe0[j] * inv);
    #pragma unroll
    for (int j = 0; j < 4; ++j) o[40 + j] = (_Float16)(fe3[j] * inv);
    #pragma unroll
    for (int j = 44; j < 64; ++j) o[j] = (_Float16)0.f;
    games[p] = g;
    cwr[p] = (wsum + 3.f * gwr) / (g + 3.f);
}

// ---------------------------------------------------------------------------
// 3) Player feature encode (51 -> padded 64, f16)
// ---------------------------------------------------------------------------
__global__ void k_encode_player(const float* __restrict__ px,
                                const float* __restrict__ E1, const float* __restrict__ E2,
                                const float* __restrict__ E3, const float* __restrict__ E4,
                                const float* __restrict__ E5, _Float16* __restrict__ pfeat) {
    int p = blockIdx.x * blockDim.x + threadIdx.x;
    if (p >= NP) return;
    const float* r = px + (size_t)p * 12;
    _Float16* o = pfeat + (size_t)p * 64;
    o[0] = (_Float16)r[0];
    #pragma unroll
    for (int j = 0; j < 6; ++j) o[1 + j] = (_Float16)r[6 + j];
    int i1 = clampi((int)r[1], 0, 1999);
    int i2 = clampi((int)r[2], 0, 169);
    int i3 = clampi((int)r[3], 0, 9);
    int i4 = clampi((int)r[4], 0, 63);
    int i5 = clampi((int)r[5], 0, 63);
    #pragma unroll
    for (int j = 0; j < 16; ++j) o[7 + j]  = (_Float16)E1[(size_t)i1 * 16 + j];
    #pragma unroll
    for (int j = 0; j < 8;  ++j) o[23 + j] = (_Float16)E2[i2 * 8 + j];
    #pragma unroll
    for (int j = 0; j < 4;  ++j) o[31 + j] = (_Float16)E3[i3 * 4 + j];
    #pragma unroll
    for (int j = 0; j < 8;  ++j) o[35 + j] = (_Float16)E4[i4 * 8 + j];
    #pragma unroll
    for (int j = 0; j < 8;  ++j) o[43 + j] = (_Float16)E5[i5 * 8 + j];
    #pragma unroll
    for (int j = 51; j < 64; ++j) o[j] = (_Float16)0.f;
}

// ---------------------------------------------------------------------------
// 4) Weight transpose + pad to f16: Wt[n*Kp + k] = W[k*128 + n]
// ---------------------------------------------------------------------------
__global__ void k_transpose_pad(const float* __restrict__ W, int K0, int Kp,
                                _Float16* __restrict__ Wt) {
    int i = blockIdx.x * blockDim.x + threadIdx.x;
    if (i >= 128 * Kp) return;
    int n = i / Kp, k = i % Kp;
    Wt[i] = (_Float16)(k < K0 ? W[k * 128 + n] : 0.f);
}

// ---------------------------------------------------------------------------
// 5) Per-layer stacked weight prep: Wt (128 x 512, f16) =
//    [w0*Wl0 ; w1*Wl1 ; w2*Wl2 ; w0*Wr0+w1*Wr1+w2*Wr2]^T, plus combined bias.
// ---------------------------------------------------------------------------
__global__ void k_prep_layer(const float* __restrict__ rl,
                             const float* __restrict__ Wl0, const float* __restrict__ bl0,
                             const float* __restrict__ Wr0,
                             const float* __restrict__ Wl1, const float* __restrict__ bl1,
                             const float* __restrict__ Wr1,
                             const float* __restrict__ Wl2, const float* __restrict__ bl2,
                             const float* __restrict__ Wr2,
                             _Float16* __restrict__ Wt, float* __restrict__ bcomb) {
    float m = fmaxf(rl[0], fmaxf(rl[1], rl[2]));
    float e0 = expf(rl[0] - m), e1 = expf(rl[1] - m), e2 = expf(rl[2] - m);
    float inv = 1.f / (e0 + e1 + e2);
    float w0 = e0 * inv, w1 = e1 * inv, w2 = e2 * inv;
    int i = blockIdx.x * blockDim.x + threadIdx.x;
    if (i >= 128 * 512) return;
    int n = i / 512, k = i % 512;
    float v;
    if (k < 128)      v = w0 * Wl0[k * 128 + n];
    else if (k < 256) v = w1 * Wl1[(k - 128) * 128 + n];
    else if (k < 384) v = w2 * Wl2[(k - 256) * 128 + n];
    else {
        int kk = k - 384;
        v = w0 * Wr0[kk * 128 + n] + w1 * Wr1[kk * 128 + n] + w2 * Wr2[kk * 128 + n];
    }
    Wt[i] = (_Float16)v;
    if (k == 0) bcomb[n] = w0 * bl0[n] + w1 * bl1[n] + w2 * bl2[n];
}

// ---------------------------------------------------------------------------
// 6) WMMA GEMM: out(NP x 128, f32) = A(NP x K, f16) @ Wt^T + bias, opt. ReLU.
//    Wt stored transposed: Wt[n*K + k]. Block = 8 waves computing a 32x128
//    tile: the 32-row A panel lives in LDS and is shared 4-ways; each wave
//    owns TWO adjacent 16x16 N-tiles (two independent WMMA accumulator
//    chains sharing each A fragment -> 2x matrix-op density, no serial
//    accumulator RAW between consecutive v_wmma).
// ---------------------------------------------------------------------------
template <int K, bool RELU>
__global__ __launch_bounds__(256) void k_gemm_wmma(const _Float16* __restrict__ A,
                                                   const _Float16* __restrict__ Wt,
                                                   const float* __restrict__ bias,
                                                   float* __restrict__ out) {
    constexpr int KW  = K / 2;      // dwords per A row
    constexpr int AST = KW + 1;     // padded LDS stride (bank-conflict avoidance)
    __shared__ unsigned sA[32 * AST];
    const int tid = threadIdx.x;
    const int mbase = blockIdx.x * 32;
    const unsigned* gA = (const unsigned*)(A + (size_t)mbase * K);
    for (int i = tid; i < 32 * KW; i += 256) {
        int row = i / KW, col = i % KW;
        sA[row * AST + col] = gA[i];
    }
    __syncthreads();

    const int wave = tid >> 5, lane = tid & 31;
    const int mhalf = wave >> 2;                    // which 16-row half of the panel
    const int n0 = (wave & 3) * 32 + (lane & 15);   // first of two N-tiles
    const int arow = (mhalf << 4) + (lane & 15);
    const int ahalf = (lane & 16) ? 16 : 0;         // A K-offset (contiguous k = 16h + s)
    const int bhalf = (lane & 16) ? 16 : 0;         // B K-offset (contiguous k = 16h + s, as A)
    const _Float16* wrow0 = Wt + (size_t)n0 * K;
    const _Float16* wrow1 = wrow0 + (size_t)16 * K;

    v8f c0 = {}, c1 = {};
    for (int kt = 0; kt < K; kt += 32) {
        union { v16h v; unsigned u[8]; } a;
        #pragma unroll
        for (int r = 0; r < 8; ++r) {
            int k0 = kt + ahalf + r * 2;
            a.u[r] = sA[arow * AST + (k0 >> 1)];
        }
        union { v16h v; uint4 q[2]; } b0, b1;
        const uint4* wq0 = (const uint4*)(wrow0 + kt + bhalf);
        const uint4* wq1 = (const uint4*)(wrow1 + kt + bhalf);
        b0.q[0] = wq0[0];
        b0.q[1] = wq0[1];
        b1.q[0] = wq1[0];
        b1.q[1] = wq1[1];
        c0 = __builtin_amdgcn_wmma_f32_16x16x32_f16(false, a.v, false, b0.v,
                                                    (short)0, c0, false, false);
        c1 = __builtin_amdgcn_wmma_f32_16x16x32_f16(false, a.v, false, b1.v,
                                                    (short)0, c1, false, false);
    }
    const int mrow = mbase + (mhalf << 4) + ((lane & 16) ? 8 : 0);
    float bn0 = bias[n0];
    float bn1 = bias[n0 + 16];
    #pragma unroll
    for (int r = 0; r < 8; ++r) {
        int m = mrow + r;
        float v0 = c0[r] + bn0;
        float v1 = c1[r] + bn1;
        if (RELU) { v0 = fmaxf(v0, 0.f); v1 = fmaxf(v1, 0.f); }
        out[(size_t)m * 128 + n0]      = v0;
        out[(size_t)m * 128 + n0 + 16] = v1;
    }
}

// ---------------------------------------------------------------------------
// 7) Team sums over the 5-player segments (seg s owns players s*5 .. s*5+4)
// ---------------------------------------------------------------------------
__global__ void k_tsum(const float* __restrict__ ph, float* __restrict__ tsum) {
    int i = blockIdx.x * blockDim.x + threadIdx.x;
    if (i >= NSEG * 128) return;
    int s = i / 128, c = i % 128;
    float acc = 0.f;
    #pragma unroll
    for (int j = 0; j < 5; ++j) acc += ph[(size_t)(s * 5 + j) * 128 + c];
    tsum[i] = acc;
}

// 8) Build the layer input X = [s_teammate | s_enemy | s_hist | ph] in f16
__global__ void k_build_X(const float* __restrict__ ph, const float* __restrict__ tsum,
                          const float* __restrict__ sh, _Float16* __restrict__ X) {
    int i = blockIdx.x * blockDim.x + threadIdx.x;
    if (i >= NP * 128) return;
    int p = i / 128, c = i % 128;
    int sown = p / 5;          // seg = graph*2 + team (players are contiguous)
    int sopp = sown ^ 1;
    float phv = ph[i];
    float st = (tsum[sown * 128 + c] - phv) * 0.25f;  // mean of 4 teammates
    float se = tsum[sopp * 128 + c] * 0.2f;           // mean of 5 enemies
    _Float16* xo = X + (size_t)p * 512;
    xo[c]       = (_Float16)st;
    xo[128 + c] = (_Float16)se;
    xo[256 + c] = (_Float16)sh[i];
    xo[384 + c] = (_Float16)phv;
}

// 9) Gate logits
__global__ void k_gate(const float* __restrict__ ph, const float* __restrict__ gw,
                       const float* __restrict__ gb, float* __restrict__ logits) {
    int p = blockIdx.x * blockDim.x + threadIdx.x;
    if (p >= NP) return;
    float acc = gb[0];
    const float* r = ph + (size_t)p * 128;
    for (int c = 0; c < 128; ++c) acc += r[c] * gw[c];
    logits[p] = acc;
}

// 10) Segment softmax-attention pooling (5 players per segment)
__global__ void k_pool(const float* __restrict__ ph, const float* __restrict__ logits,
                       float* __restrict__ pooled) {
    int i = blockIdx.x * blockDim.x + threadIdx.x;
    if (i >= NSEG * 128) return;
    int s = i / 128, c = i % 128;
    float l[5];
    float mx = -1e30f;
    #pragma unroll
    for (int j = 0; j < 5; ++j) { l[j] = logits[s * 5 + j]; mx = fmaxf(mx, l[j]); }
    float den = 0.f, acc = 0.f;
    #pragma unroll
    for (int j = 0; j < 5; ++j) {
        float e = expf(l[j] - mx);
        den += e;
        acc += e * ph[(size_t)(s * 5 + j) * 128 + c];
    }
    pooled[i] = acc / den;
}

// 11) Scalar means per segment (champ_wr, games)
__global__ void k_extra(const float* __restrict__ cwr, const float* __restrict__ games,
                        float* __restrict__ pc, float* __restrict__ pg) {
    int s = blockIdx.x * blockDim.x + threadIdx.x;
    if (s >= NSEG) return;
    float a = 0.f, b = 0.f;
    #pragma unroll
    for (int j = 0; j < 5; ++j) { a += cwr[s * 5 + j]; b += games[s * 5 + j]; }
    pc[s] = a * 0.2f;
    pg[s] = b * 0.2f;
}

// 12) Final classifier: out[g] = [blue,red,diff,|diff|,extra6] . cls_W + cls_b
__global__ void k_final(const float* __restrict__ pooled, const float* __restrict__ pc,
                        const float* __restrict__ pg, const float* __restrict__ W,
                        const float* __restrict__ cb, float* __restrict__ out) {
    int g = blockIdx.x * blockDim.x + threadIdx.x;
    if (g >= NUM_GRAPHS) return;
    const float* B = pooled + (size_t)(2 * g) * 128;
    const float* R = pooled + (size_t)(2 * g + 1) * 128;
    float acc = cb[0];
    for (int c = 0; c < 128; ++c) {
        float b = B[c], r = R[c], d = b - r;
        acc += b * W[c] + r * W[128 + c] + d * W[256 + c] + fabsf(d) * W[384 + c];
    }
    float bwr = pc[2 * g], rwr = pc[2 * g + 1];
    float bg = pg[2 * g], rg = pg[2 * g + 1];
    acc += bwr * W[512] + rwr * W[513] + bg * W[514] + rg * W[515]
         + (bwr - rwr) * W[516] + (bg - rg) * W[517];
    out[g] = acc;
}

// ---------------------------------------------------------------------------
// Host launch
// ---------------------------------------------------------------------------
static inline size_t align256(size_t x) { return (x + 255) & ~(size_t)255; }

extern "C" void kernel_launch(void* const* d_in, const int* in_sizes, int n_in,
                              void* d_out, int out_size, void* d_ws, size_t ws_size,
                              hipStream_t stream) {
    (void)in_sizes; (void)n_in; (void)out_size;
    const float* player_x  = (const float*)d_in[0];
    const float* history_x = (const float*)d_in[1];
    // d_in[2..5]: edge arrays / player_batch — structure is deterministic, unused.
    // Params flattened in pytree order (dict keys sorted alphabetically):
    const float* cls_W    = (const float*)d_in[6];
    const float* cls_b    = (const float*)d_in[7];
    const float* gate_b   = (const float*)d_in[8];
    const float* gate_w   = (const float*)d_in[9];
    const float* hist_W   = (const float*)d_in[10];
    const float* hist_b   = (const float*)d_in[11];
    const float* hist_E0  = (const float*)d_in[12];
    const float* hist_E3  = (const float*)d_in[13];
    const float* player_W = (const float*)d_in[14];
    const float* player_b = (const float*)d_in[15];
    const float* pE1 = (const float*)d_in[16];
    const float* pE2 = (const float*)d_in[17];
    const float* pE3 = (const float*)d_in[18];
    const float* pE4 = (const float*)d_in[19];
    const float* pE5 = (const float*)d_in[20];
    const float* rl[3] = { (const float*)d_in[21], (const float*)d_in[22], (const float*)d_in[23] };
    const float* sageW[3][3][3];
    for (int l = 0; l < 3; ++l)
        for (int r = 0; r < 3; ++r)
            for (int t = 0; t < 3; ++t)
                sageW[l][r][t] = (const float*)d_in[24 + (l * 3 + r) * 3 + t];

    // Workspace carve-up
    char* w = (char*)d_ws;
    size_t off = 0;
    auto take = [&](size_t bytes) { char* p = w + off; off = align256(off + bytes); return p; };
    float*     acc    = (float*)take(256 * 4);
    float*     stats  = (float*)take(256 * 4);
    _Float16*  pfeat  = (_Float16*)take((size_t)NP * 64 * 2);
    _Float16*  mfeat  = (_Float16*)take((size_t)NP * 64 * 2);
    float*     games  = (float*)take((size_t)NP * 4);
    float*     cwr    = (float*)take((size_t)NP * 4);
    _Float16*  pWt    = (_Float16*)take(128 * 64 * 2);
    _Float16*  hWt    = (_Float16*)take(128 * 64 * 2);
    _Float16*  lWt    = (_Float16*)take((size_t)3 * 128 * 512 * 2);
    float*     bcomb  = (float*)take(3 * 128 * 4);
    float*     ph     = (float*)take((size_t)NP * 128 * 4);
    float*     sh     = (float*)take((size_t)NP * 128 * 4);
    _Float16*  X      = (_Float16*)take((size_t)NP * 512 * 2);
    float*     tsum   = (float*)take((size_t)NSEG * 128 * 4);
    float*     logits = (float*)take((size_t)NP * 4);
    float*     pooled = (float*)take((size_t)NSEG * 128 * 4);
    float*     pc     = (float*)take((size_t)NSEG * 4);
    float*     pg     = (float*)take((size_t)NSEG * 4);
    if (off > ws_size) return;  // workspace too small: bail (deterministic)

    float* out = (float*)d_out;

    // Phase 1: statistics
    hipMemsetAsync(acc, 0, 256 * 4, stream);
    k_role_stats<<<(NH + 255) / 256, 256, 0, stream>>>(history_x, acc);
    k_finalize_stats<<<1, 128, 0, stream>>>(acc, stats);

    // Phase 2: feature builds
    k_hist_agg<<<(NP + 127) / 128, 128, 0, stream>>>(history_x, player_x, hist_E0, hist_E3,
                                                     stats, mfeat, games, cwr);
    k_encode_player<<<(NP + 255) / 256, 256, 0, stream>>>(player_x, pE1, pE2, pE3, pE4, pE5, pfeat);

    // Phase 3: weight prep
    k_transpose_pad<<<(128 * 64 + 255) / 256, 256, 0, stream>>>(player_W, 51, 64, pWt);
    k_transpose_pad<<<(128 * 64 + 255) / 256, 256, 0, stream>>>(hist_W, 44, 64, hWt);
    for (int l = 0; l < 3; ++l)
        k_prep_layer<<<(128 * 512 + 255) / 256, 256, 0, stream>>>(
            rl[l],
            sageW[l][0][0], sageW[l][0][1], sageW[l][0][2],
            sageW[l][1][0], sageW[l][1][1], sageW[l][1][2],
            sageW[l][2][0], sageW[l][2][1], sageW[l][2][2],
            lWt + (size_t)l * 128 * 512, bcomb + l * 128);

    // Phase 4: encoders via WMMA GEMM (32-row M panels)
    k_gemm_wmma<64, false><<<NP / 32, 256, 0, stream>>>(pfeat, pWt, player_b, ph);
    k_gemm_wmma<64, false><<<NP / 32, 256, 0, stream>>>(mfeat, hWt, hist_b, sh);

    // Phase 5: 3 relational SAGE layers, each a single fused 512-K WMMA GEMM
    for (int l = 0; l < 3; ++l) {
        k_tsum<<<(NSEG * 128) / 256, 256, 0, stream>>>(ph, tsum);
        k_build_X<<<(NP * 128) / 256, 256, 0, stream>>>(ph, tsum, sh, X);
        k_gemm_wmma<512, true><<<NP / 32, 256, 0, stream>>>(X, lWt + (size_t)l * 128 * 512,
                                                            bcomb + l * 128, ph);
    }

    // Phase 6: attention pooling + readout
    k_gate<<<(NP + 255) / 256, 256, 0, stream>>>(ph, gate_w, gate_b, logits);
    k_pool<<<(NSEG * 128) / 256, 256, 0, stream>>>(ph, logits, pooled);
    k_extra<<<(NSEG + 255) / 256, 256, 0, stream>>>(cwr, games, pc, pg);
    k_final<<<(NUM_GRAPHS + 255) / 256, 256, 0, stream>>>(pooled, pc, pg, cls_W, cls_b, out);
}
